// SingleHeadAttention_1090921693422
// MI455X (gfx1250) — matrix operand order repaired and checked by reading, not compiled, not run on hardware
//
#include <hip/hip_runtime.h>
#include <hip/hip_bf16.h>
#include <math.h>

typedef _Float16 half16 __attribute__((ext_vector_type(16)));
typedef _Float16 half8  __attribute__((ext_vector_type(8)));
typedef float    f32x8  __attribute__((ext_vector_type(8)));
typedef float    f32x4  __attribute__((ext_vector_type(4)));

#define Bsz 8
#define Tsz 2048
#define Csz 1024
#define Hsz 64

// ---------------------------------------------------------------------------
// Kernel 1: W[C,H] f32  ->  Wt[H,C] f16  (so WMMA B-operand loads are contiguous)
// ---------------------------------------------------------------------------
__global__ __launch_bounds__(256) void prep_weights_kernel(
    const float* __restrict__ Wq, const float* __restrict__ Wk,
    const float* __restrict__ Wv,
    _Float16* __restrict__ Wtq, _Float16* __restrict__ Wtk,
    _Float16* __restrict__ Wtv)
{
    int idx = blockIdx.x * 256 + threadIdx.x;          // 0 .. C*H-1
    if (idx < Csz * Hsz) {
        int c = idx >> 6;        // 0..1023
        int h = idx & 63;        // 0..63
        Wtq[h * Csz + c] = (_Float16)Wq[c * Hsz + h];
        Wtk[h * Csz + c] = (_Float16)Wk[c * Hsz + h];
        Wtv[h * Csz + c] = (_Float16)Wv[c * Hsz + h];
    }
}

// ---------------------------------------------------------------------------
// Kernel 2: QKV projection.  One wave handles a 16-row tile of X.
// Weight chunks (3 matrices x 64 rows x 32 k = 12KB) are staged in LDS with
// double-buffered gfx1250 async-to-LDS copies (ASYNCcnt), shared by 4 waves.
// Produces Q,K row-major f16 [B,T,H] and V transposed f16 [B,H,T].
// ---------------------------------------------------------------------------
__global__ __launch_bounds__(128) void qkv_proj_kernel(
    const float* __restrict__ X,
    const _Float16* __restrict__ Wtq, const _Float16* __restrict__ Wtk,
    const _Float16* __restrict__ Wtv,
    _Float16* __restrict__ Qf, _Float16* __restrict__ Kf,
    _Float16* __restrict__ Vt)
{
    // [buf][matrix][n(64)][k(32)] f16 = 2 * 12KB
    __shared__ __attribute__((aligned(32))) _Float16 Wlds[2][3][64][32];

    const int lane = threadIdx.x & 31;
    const int wave = threadIdx.x >> 5;
    const int tid  = threadIdx.x;
    const int tile = blockIdx.x * 4 + wave;   // 0 .. B*T/16-1 = 1023
    const int b    = tile >> 7;               // 128 tiles per batch
    const int t0   = (tile & 127) * 16;       // base row in T
    const int m    = lane & 15;               // A-layout: lane%16 = M row
    const int g    = lane >> 4;               // lane group (K-half select)

    const float* __restrict__ xrow = X + ((size_t)b * Tsz + t0 + m) * Csz;
    const _Float16* wsrc[3] = {Wtq, Wtk, Wtv};

    // cooperative async copy of one 12KB weight chunk: 768 x b128, 6 per thread
    auto issue_chunk = [&](int buf, int kb) {
#pragma unroll
        for (int j = 0; j < 6; j++) {
            const int c   = j * 128 + tid;        // 0..767
            const int mat = c >> 8;               // 0..2
            const int rem = c & 255;
            const int n   = rem >> 2;             // 0..63
            const int q   = rem & 3;              // 16B quarter of a 64B row
            const _Float16* src = wsrc[mat] + (size_t)n * Csz + kb + q * 8;
            const unsigned  dst = (unsigned)(uintptr_t)(void*)&Wlds[buf][mat][n][q * 8];
            const unsigned long long ga = (unsigned long long)(uintptr_t)(const void*)src;
            __asm__ volatile("global_load_async_to_lds_b128 %0, %1, off"
                             :: "v"(dst), "v"(ga) : "memory");
        }
    };

    f32x8 zero = {};
    f32x8 accQ[4], accK[4], accV[4];
#pragma unroll
    for (int t = 0; t < 4; t++) { accQ[t] = zero; accK[t] = zero; accV[t] = zero; }

    issue_chunk(0, 0);

    for (int kb = 0, it = 0; kb < Csz; kb += 32, ++it) {
        const int cur = it & 1;
        if (kb + 32 < Csz) {
            issue_chunk(cur ^ 1, kb + 32);                       // prefetch next chunk
            __asm__ volatile("s_wait_asynccnt 0x6" ::: "memory"); // current chunk landed
            __builtin_prefetch(xrow + kb + 32, 0, 1);             // next A chunk -> caches
        } else {
            __asm__ volatile("s_wait_asynccnt 0x0" ::: "memory");
        }
        __syncthreads();   // chunk `cur` visible to all 4 waves

        // ---- A operand: X[m, kb..kb+31] in 16-bit A layout ----
        half16 a;
        const f32x4 x0 = *(const f32x4*)(xrow + kb + 16 * g);
        const f32x4 x1 = *(const f32x4*)(xrow + kb + 16 * g + 4);
        const f32x4 x2 = *(const f32x4*)(xrow + kb + 16 * g + 8);
        const f32x4 x3 = *(const f32x4*)(xrow + kb + 16 * g + 12);
#pragma unroll
        for (int i = 0; i < 4; i++) {
            a[i]      = (_Float16)x0[i];
            a[4 + i]  = (_Float16)x1[i];
            a[8 + i]  = (_Float16)x2[i];
            a[12 + i] = (_Float16)x3[i];
        }

        // ---- B operands from LDS (ds_load_b128 x2 each) ----
#pragma unroll
        for (int t = 0; t < 4; t++) {
            const half16 bq = *(const half16*)&Wlds[cur][0][16 * t + m][16 * g];
            const half16 bk = *(const half16*)&Wlds[cur][1][16 * t + m][16 * g];
            const half16 bv = *(const half16*)&Wlds[cur][2][16 * t + m][16 * g];
            accQ[t] = __builtin_amdgcn_wmma_f32_16x16x32_f16(false, a, false, bq,
                                                             (short)0, accQ[t], false, false);
            accK[t] = __builtin_amdgcn_wmma_f32_16x16x32_f16(false, a, false, bk,
                                                             (short)0, accK[t], false, false);
            accV[t] = __builtin_amdgcn_wmma_f32_16x16x32_f16(false, a, false, bv,
                                                             (short)0, accV[t], false, false);
        }
        __syncthreads();   // everyone done reading `cur` before it is overwritten
    }

    // D layout: VGPR r holds M = r + 8*g, N = lane%16
#pragma unroll
    for (int t = 0; t < 4; t++) {
#pragma unroll
        for (int r = 0; r < 8; r++) {
            const int row = t0 + r + 8 * g;
            const int h   = 16 * t + m;
            Qf[((size_t)b * Tsz + row) * Hsz + h] = (_Float16)accQ[t][r];
            Kf[((size_t)b * Tsz + row) * Hsz + h] = (_Float16)accK[t][r];
            Vt[((size_t)b * Hsz + h) * Tsz + row] = (_Float16)accV[t][r];
        }
    }
}

// ---------------------------------------------------------------------------
// Kernel 3: flash attention.  One wave per 16-query tile, online softmax
// over 32-key blocks up to the causal boundary.  Per-wave trip counts ->
// no block barriers; per-wave LDS ordering via s_wait_dscnt.
// ---------------------------------------------------------------------------
__global__ __launch_bounds__(128) void attn_kernel(
    const _Float16* __restrict__ Qf, const _Float16* __restrict__ Kf,
    const _Float16* __restrict__ Vt, float* __restrict__ out)
{
    __shared__ __attribute__((aligned(32))) _Float16 Pbuf[4][16 * 32];

    const int lane = threadIdx.x & 31;
    const int wave = threadIdx.x >> 5;
    const int tile = blockIdx.x * 4 + wave;   // 0..1023
    const int b    = tile >> 7;
    const int q0   = (tile & 127) * 16;
    const int m    = lane & 15;
    const int g    = lane >> 4;

    // ---- preload Q A-tiles for both 32-wide h chunks ----
    half16 aQ[2];
    const _Float16* __restrict__ qrow = Qf + ((size_t)b * Tsz + q0 + m) * Hsz;
#pragma unroll
    for (int c = 0; c < 2; c++) {
        const half8 lo = *(const half8*)(qrow + 32 * c + 16 * g);
        const half8 hi = *(const half8*)(qrow + 32 * c + 16 * g + 8);
#pragma unroll
        for (int i = 0; i < 8; i++) { aQ[c][i] = lo[i]; aQ[c][8 + i] = hi[i]; }
    }

    f32x8 zero = {};
    f32x8 O[4] = {zero, zero, zero, zero};
    float rowM[8], rowL[8];
#pragma unroll
    for (int r = 0; r < 8; r++) { rowM[r] = -INFINITY; rowL[r] = 0.0f; }

    const _Float16* __restrict__ Kb = Kf + (size_t)b * Tsz * Hsz;
    const _Float16* __restrict__ Vb = Vt + (size_t)b * Hsz * Tsz;
    _Float16* __restrict__ pb = &Pbuf[wave][0];

    const int kbEnd = (q0 + 15) >> 5;          // last 32-key block (inclusive)
    for (int kb = 0; kb <= kbEnd; kb++) {
        const int k0 = kb * 32;

        // ---- S = Q K^T for two 16-key subtiles ----
        f32x8 S[2];
#pragma unroll
        for (int s = 0; s < 2; s++) {
            const _Float16* krow = Kb + (size_t)(k0 + s * 16 + m) * Hsz;
            const half16 b0 = *(const half16*)(krow + 16 * g);       // h chunk 0
            const half16 b1 = *(const half16*)(krow + 32 + 16 * g);  // h chunk 1
            f32x8 t = __builtin_amdgcn_wmma_f32_16x16x32_f16(false, aQ[0], false, b0,
                                                             (short)0, zero, false, false);
            S[s]    = __builtin_amdgcn_wmma_f32_16x16x32_f16(false, aQ[1], false, b1,
                                                             (short)0, t, false, false);
        }

        // ---- causal mask + online softmax (per D-layout row r) ----
        float p0[8], p1[8];
#pragma unroll
        for (int r = 0; r < 8; r++) {
            const int qg = q0 + r + 8 * g;
            float s0 = S[0][r] * 0.125f;                 // 1/sqrt(64)
            float s1 = S[1][r] * 0.125f;
            if (k0 + m > qg)      s0 = -INFINITY;
            if (k0 + 16 + m > qg) s1 = -INFINITY;
            float bm = fmaxf(s0, s1);
            bm = fmaxf(bm, __shfl_xor(bm, 1, 32));
            bm = fmaxf(bm, __shfl_xor(bm, 2, 32));
            bm = fmaxf(bm, __shfl_xor(bm, 4, 32));
            bm = fmaxf(bm, __shfl_xor(bm, 8, 32));
            const float mn = fmaxf(rowM[r], bm);
            const float al = __expf(rowM[r] - mn);
            p0[r] = __expf(s0 - mn);
            p1[r] = __expf(s1 - mn);
            float sum = p0[r] + p1[r];
            sum += __shfl_xor(sum, 1, 32);
            sum += __shfl_xor(sum, 2, 32);
            sum += __shfl_xor(sum, 4, 32);
            sum += __shfl_xor(sum, 8, 32);
            rowL[r] = rowL[r] * al + sum;
            rowM[r] = mn;
#pragma unroll
            for (int t = 0; t < 4; t++) O[t][r] *= al;   // rescale O accumulators
        }

        // ---- transpose P from D layout to A layout through LDS ----
#pragma unroll
        for (int r = 0; r < 8; r++) {
            const int row = r + 8 * g;
            pb[row * 32 + m]      = (_Float16)p0[r];
            pb[row * 32 + 16 + m] = (_Float16)p1[r];
        }
        __asm__ volatile("s_wait_dscnt 0" ::: "memory");   // per-wave LDS ordering
        half16 aP;
        {
            const half8 lo = *(const half8*)(pb + m * 32 + 16 * g);
            const half8 hi = *(const half8*)(pb + m * 32 + 16 * g + 8);
#pragma unroll
            for (int i = 0; i < 8; i++) { aP[i] = lo[i]; aP[8 + i] = hi[i]; }
        }

        // ---- O += P V  (V pre-transposed: contiguous key dim) ----
#pragma unroll
        for (int t = 0; t < 4; t++) {
            const half16 bv = *(const half16*)(Vb + (size_t)(16 * t + m) * Tsz + k0 + 16 * g);
            O[t] = __builtin_amdgcn_wmma_f32_16x16x32_f16(false, aP, false, bv,
                                                          (short)0, O[t], false, false);
        }
    }

    // ---- epilogue: divide by softmax denom, store fp32 ----
#pragma unroll
    for (int t = 0; t < 4; t++) {
#pragma unroll
        for (int r = 0; r < 8; r++) {
            const int row = q0 + r + 8 * g;
            out[((size_t)b * Tsz + row) * Hsz + 16 * t + m] = O[t][r] / rowL[r];
        }
    }
}

// ---------------------------------------------------------------------------
extern "C" void kernel_launch(void* const* d_in, const int* in_sizes, int n_in,
                              void* d_out, int out_size, void* d_ws, size_t ws_size,
                              hipStream_t stream) {
    (void)in_sizes; (void)n_in; (void)out_size; (void)ws_size;

    const float* X  = (const float*)d_in[0];
    const float* Wq = (const float*)d_in[1];
    const float* Wk = (const float*)d_in[2];
    const float* Wv = (const float*)d_in[3];
    float* out = (float*)d_out;

    char* ws = (char*)d_ws;
    const size_t WT_BYTES = (size_t)Hsz * Csz * sizeof(_Float16);        // 128 KB each
    const size_t QK_BYTES = (size_t)Bsz * Tsz * Hsz * sizeof(_Float16);  // 2 MB each
    _Float16* Wtq = (_Float16*)(ws);
    _Float16* Wtk = (_Float16*)(ws + WT_BYTES);
    _Float16* Wtv = (_Float16*)(ws + 2 * WT_BYTES);
    _Float16* Qf  = (_Float16*)(ws + 3 * WT_BYTES);
    _Float16* Kf  = (_Float16*)(ws + 3 * WT_BYTES + QK_BYTES);
    _Float16* Vt  = (_Float16*)(ws + 3 * WT_BYTES + 2 * QK_BYTES);

    prep_weights_kernel<<<(Csz * Hsz + 255) / 256, 256, 0, stream>>>(
        Wq, Wk, Wv, Wtq, Wtk, Wtv);

    // B*T/16 = 1024 wave-tiles, 4 waves per block
    qkv_proj_kernel<<<256, 128, 0, stream>>>(X, Wtq, Wtk, Wtv, Qf, Kf, Vt);

    attn_kernel<<<256, 128, 0, stream>>>(Qf, Kf, Vt, out);
}
